// XModel_27608049779426
// MI455X (gfx1250) — matrix operand order repaired and checked by reading, not compiled, not run on hardware
//
#include <hip/hip_runtime.h>
#include <hip/hip_bf16.h>
#include <math.h>

typedef __attribute__((ext_vector_type(16))) _Float16 v16h;
typedef __attribute__((ext_vector_type(8)))  float    v8f;
typedef _Float16 half_t;

#define CDIV(a,b) (((a)+(b)-1)/(b))

// sizes
#define HW        96
#define NPIX      9216        // 96*96
#define PWW       102
#define PWPIX     10404       // 102*102
#define PWPAD     10416       // 651*16 (padded N for WMMA tiling)
#define C64       64
#define C256      256

static __device__ inline v8f wmma_f16(v16h a, v16h b, v8f c) {
  // D = A(16x32 f16) * B(32x16 f16) + C(16x16 f32)
  return __builtin_amdgcn_wmma_f32_16x16x32_f16(false, a, false, b, (short)0, c,
                                                false, false);
}

// ---------------------------------------------------------------- DWT
__global__ void k_dwt(const float* __restrict__ img, float* __restrict__ ll,
                      float* __restrict__ lh, float* __restrict__ hl,
                      float* __restrict__ hh) {
  int t = blockIdx.x * blockDim.x + threadIdx.x;
  if (t >= NPIX) return;
  const float S = 0.70710678118654752f;
  int a = t / HW, b = t % HW;
  float x00 = img[(2 * a) * 192 + 2 * b];
  float x10 = img[(2 * a + 1) * 192 + 2 * b];
  float x01 = img[(2 * a) * 192 + 2 * b + 1];
  float x11 = img[(2 * a + 1) * 192 + 2 * b + 1];
  float lo0 = (x00 + x10) * S, lo1 = (x01 + x11) * S;
  float hi0 = (x00 - x10) * S, hi1 = (x01 - x11) * S;
  ll[t] = (lo0 + lo1) * S;
  lh[t] = (lo0 - lo1) * S;
  hl[t] = (hi0 + hi1) * S;
  hh[t] = (hi0 - hi1) * S;
}

// ------------------------------------------------------- IDWT + residual
__global__ void k_idwt(const float* __restrict__ LL, const float* __restrict__ LH,
                       const float* __restrict__ HL, const float* __restrict__ HH,
                       const float* __restrict__ img, float* __restrict__ out) {
  int t = blockIdx.x * blockDim.x + threadIdx.x;
  if (t >= 192 * 192) return;
  int i = t / 192, j = t % 192;
  int a = i >> 1, p = i & 1, b = j >> 1, q = j & 1;
  float ll = LL[a * HW + b], lh = LH[a * HW + b];
  float hl = HL[a * HW + b], hh = HH[a * HW + b];
  float sq = q ? -1.f : 1.f, sp = p ? -1.f : 1.f;
  float v = 0.5f * ((ll + sq * lh) + sp * (hl + sq * hh));
  out[t] = v + img[t];
}

// --------------------------------------- ip: 1->64 3x3 conv (with transpose)
// F is channels-last [u][v][c] f32
__global__ void k_ip(const float* __restrict__ sb, const float* __restrict__ w,
                     const float* __restrict__ bi, float* __restrict__ F) {
  int t = blockIdx.x * blockDim.x + threadIdx.x;
  if (t >= NPIX * C64) return;
  int c = t & 63, p = t >> 6;
  int u = p / HW, v = p % HW;
  float acc = bi[c];
#pragma unroll
  for (int dy = 0; dy < 3; ++dy)
#pragma unroll
    for (int dx = 0; dx < 3; ++dx) {
      int q = v + dx - 1, r = u + dy - 1;   // input is sb transposed: sb[q][r]
      if ((unsigned)q < HW && (unsigned)r < HW)
        acc += w[c * 9 + dy * 3 + dx] * sb[q * HW + r];
    }
  F[t] = acc;
}

// ---------------------------------------- op: 64->1 3x3 (flipped+transposed)
__global__ void k_op(const float* __restrict__ X, const float* __restrict__ w,
                     const float* __restrict__ b, float* __restrict__ out) {
  int t = blockIdx.x * blockDim.x + threadIdx.x;
  if (t >= NPIX) return;
  int a = t / HW, bb = t % HW;
  float acc = b[0];
  for (int dy = 0; dy < 3; ++dy)
    for (int dx = 0; dx < 3; ++dx) {
      int u = bb + dy - 1, v = a + dx - 1;
      if ((unsigned)u < HW && (unsigned)v < HW) {
        const float* xp = X + ((size_t)(u * HW + v)) * C64;
        for (int c = 0; c < C64; ++c)
          acc += w[c * 9 + (2 - dy) * 3 + (2 - dx)] * xp[c];
      }
    }
  out[t] = acc;
}

// ----------------------------- depthwise 7x7 (pad 6 total), out 102x102 chlast
__global__ void k_dw7(const float* __restrict__ X, const float* __restrict__ dw,
                      float* __restrict__ H32, half_t* __restrict__ H16) {
  int t = blockIdx.x * blockDim.x + threadIdx.x;
  if (t >= PWPAD * C64) return;
  int c = t & 63, p = t >> 6;
  float acc = 0.f;
  if (p < PWPIX) {
    int y = p / PWW, x = p % PWW;
    for (int ky = 0; ky < 7; ++ky) {
      int yy = y + ky - 6;
      if ((unsigned)yy < HW) {
        for (int kx = 0; kx < 7; ++kx) {
          int xx = x + kx - 6;
          if ((unsigned)xx < HW)
            acc += dw[c * 49 + ky * 7 + kx] * X[((size_t)(yy * HW + xx)) * C64 + c];
        }
      }
    }
  }
  H32[t] = acc;
  H16[t] = (half_t)acc;
}

// ------------------------------------------------- pack row-major A -> frags
// dst layout: [mt][kt][lane(32)][j(16)]; A element (m,k) at A[m*strideM + k]
__global__ void k_pack_rm(const float* __restrict__ A, int strideM,
                          half_t* __restrict__ dst, int MT, int KT) {
  int t = blockIdx.x * blockDim.x + threadIdx.x;
  int total = MT * KT * 512;
  if (t >= total) return;
  int j = t & 15, lane = (t >> 4) & 31;
  int tile = t >> 9;
  int kt = tile % KT, mt = tile / KT;
  int m = mt * 16 + (lane & 15);
  int h = lane >> 4;
  int k = kt * 32 + 16 * h + j;
  dst[t] = (half_t)A[(size_t)m * strideM + k];
}

// ------------------------------- pack l1 weights: [49][16][2][lane][j] frags
// W = cb_l1w block i: [256][64][49]
__global__ void k_pack_l1(const float* __restrict__ W, half_t* __restrict__ dst) {
  int t = blockIdx.x * blockDim.x + threadIdx.x;
  if (t >= 49 * 16 * 2 * 512) return;
  int j = t & 15, lane = (t >> 4) & 31;
  int kt = (t >> 9) & 1, mt = (t >> 10) & 15, s = t >> 14;
  int m = mt * 16 + (lane & 15);
  int h = lane >> 4;
  int c = kt * 32 + 16 * h + j;
  dst[t] = (half_t)W[((size_t)m * 64 + c) * 49 + s];
}

// -------------------------------------------------------- generic WMMA GEMM
// D[n][m] = sum_k B[n][k] * A[m][k], wave = one 16x16 tile.
// B: f16 [>=NT*16][Kdim]; apack: [MT][KT][32][16] f16.
__global__ __launch_bounds__(256) void k_gemm(
    const half_t* __restrict__ apack, const half_t* __restrict__ B,
    int MT, int NT, int KT, int Kdim, int Mtot, int Npix,
    const float* __restrict__ bias, const float* __restrict__ resid,
    const float* __restrict__ resid2, float* __restrict__ outF32,
    half_t* __restrict__ outF16, int do_gelu) {
  int wave = blockIdx.x * 8 + (threadIdx.x >> 5);
  if (wave >= MT * NT) return;     // wave-uniform exit: EXEC stays all-ones
  int lane = threadIdx.x & 31;
  int mt = wave % MT, nt = wave / MT;
  int ln = lane & 15, kg = lane >> 4;
  int n = nt * 16 + ln;
  v8f acc = {};
  const half_t* bptr = B + (size_t)n * Kdim + kg * 16;
  const half_t* aptr = apack + ((size_t)(mt * KT) * 32 + lane) * 16;
  for (int kt = 0; kt < KT; ++kt) {
    v16h a = *(const v16h*)aptr;
    v16h b = *(const v16h*)bptr;
    acc = wmma_f16(a, b, acc);
    aptr += 512;
    bptr += 32;
  }
  int mbase = mt * 16 + kg * 8;
  size_t obase = (size_t)n * Mtot + mbase;
  bool valid = (n < Npix);
#pragma unroll
  for (int r = 0; r < 8; ++r) {
    float v = 0.f;
    if (valid) {
      v = acc[r];
      int m = mbase + r;
      if (bias) v += bias[m];
      if (do_gelu) v = 0.5f * v * (1.0f + erff(v * 0.70710678118654752f));
      if (resid) v += resid[obase + r];
      if (resid2) v += resid2[obase + r];
    }
    if (outF32) outF32[obase + r] = v;
    if (outF16) outF16[obase + r] = (half_t)v;
  }
}

// ------------------- 7x7 valid conv 64->256 as 49 shifted K=64 WMMA GEMMs
// Cooperative version: one block (8 waves) owns one 32-pixel N strip; the
// 7x38x64 f16 activation tile is staged into LDS once via async-to-LDS DMA,
// then all 16 M-tiles are computed from LDS (2 M-tiles x 2 N-tiles per wave).
// Hn: f16 chlast [102][102][64]; apack: [49][16][2][32][16]; Gn: [9216][256]
#define LROW 2432                 // 38 cols * 64 ch (halfs)
__global__ __launch_bounds__(256) void k_l1conv(
    const half_t* __restrict__ apack, const half_t* __restrict__ Hn,
    const float* __restrict__ bias, half_t* __restrict__ Gn) {
  __shared__ __align__(32) half_t sB[7 * LROW];   // 34 KB
  int np = blockIdx.x;            // 0..287
  int y = np / 3;
  int x0 = (np % 3) * 32;

  // ---- stage B tile: rows y..y+6, cols x0..x0+37, 64 ch, 16B chunks
  {
    const char* gbase = (const char*)Hn + ((size_t)(y * PWW + x0)) * (C64 * 2);
    unsigned sbase = (unsigned)(size_t)(&sB[0]);
    for (int q = threadIdx.x; q < 7 * 304; q += 256) {   // 304 = 38*64*2/16
      int row = q / 304, off = q % 304;
      unsigned long long ga =
          (unsigned long long)(size_t)(gbase + (size_t)row * (PWW * C64 * 2) +
                                       (size_t)off * 16);
      unsigned la = sbase + row * (LROW * 2) + off * 16;
      asm volatile("global_load_async_to_lds_b128 %0, %1, off"
                   :: "v"(la), "v"(ga) : "memory");
    }
    asm volatile("s_wait_asynccnt 0x0" ::: "memory");
    __syncthreads();
  }

  int w = threadIdx.x >> 5;       // wave -> M-tile pair, 0..7
  int lane = threadIdx.x & 31;
  int ln = lane & 15, kg = lane >> 4;
  v8f acc00 = {}, acc01 = {}, acc10 = {}, acc11 = {};
  for (int s = 0; s < 49; ++s) {
    int ky = s / 7, kx = s % 7;
    const half_t* bl = sB + ky * LROW + (kx + ln) * C64 + kg * 16;
    const half_t* ap = apack + ((size_t)(s * 16 + 2 * w) * 64 + lane) * 16;
    // kt = 0
    v16h a0 = *(const v16h*)ap;
    v16h a1 = *(const v16h*)(ap + 1024);
    v16h b0 = *(const v16h*)bl;
    v16h b1 = *(const v16h*)(bl + 16 * C64);
    acc00 = wmma_f16(a0, b0, acc00);
    acc01 = wmma_f16(a0, b1, acc01);
    acc10 = wmma_f16(a1, b0, acc10);
    acc11 = wmma_f16(a1, b1, acc11);
    // kt = 1
    a0 = *(const v16h*)(ap + 512);
    a1 = *(const v16h*)(ap + 1024 + 512);
    b0 = *(const v16h*)(bl + 32);
    b1 = *(const v16h*)(bl + 16 * C64 + 32);
    acc00 = wmma_f16(a0, b0, acc00);
    acc01 = wmma_f16(a0, b1, acc01);
    acc10 = wmma_f16(a1, b0, acc10);
    acc11 = wmma_f16(a1, b1, acc11);
  }
  int n0 = y * HW + x0;
  int mb = (2 * w) * 16 + kg * 8;
  auto emit = [&](v8f acc, int n, int mbase) {
#pragma unroll
    for (int r = 0; r < 8; ++r) {
      int m = mbase + r;
      float v = acc[r] + bias[m];
      v = 0.5f * v * (1.0f + erff(v * 0.70710678118654752f));   // exact GELU
      Gn[(size_t)n * C256 + m] = (half_t)v;
    }
  };
  emit(acc00, n0 + ln, mb);
  emit(acc01, n0 + 16 + ln, mb);
  emit(acc10, n0 + ln, mb + 16);
  emit(acc11, n0 + 16 + ln, mb + 16);
}

// --------------------------------------------------- global mean/var reduce
__global__ void k_red1(const float* __restrict__ src, int count,
                       float* __restrict__ partial) {
  __shared__ float s1[256], s2[256];
  float a = 0.f, b = 0.f;
  for (int i = blockIdx.x * 256 + threadIdx.x; i < count; i += gridDim.x * 256) {
    float v = src[i];
    a += v;
    b += v * v;
  }
  s1[threadIdx.x] = a;
  s2[threadIdx.x] = b;
  __syncthreads();
  for (int s = 128; s > 0; s >>= 1) {
    if (threadIdx.x < s) {
      s1[threadIdx.x] += s1[threadIdx.x + s];
      s2[threadIdx.x] += s2[threadIdx.x + s];
    }
    __syncthreads();
  }
  if (threadIdx.x == 0) {
    partial[blockIdx.x * 2] = s1[0];
    partial[blockIdx.x * 2 + 1] = s2[0];
  }
}

__global__ void k_red2(const float* __restrict__ partial, float count,
                       float* __restrict__ stats) {
  __shared__ float s1[256], s2[256];
  int t = threadIdx.x;
  s1[t] = partial[2 * t];
  s2[t] = partial[2 * t + 1];
  __syncthreads();
  for (int s = 128; s > 0; s >>= 1) {
    if (t < s) {
      s1[t] += s1[t + s];
      s2[t] += s2[t + s];
    }
    __syncthreads();
  }
  if (t == 0) {
    float m = s1[0] / count;
    float var = s2[0] / count - m * m;
    stats[0] = m;
    stats[1] = rsqrtf(var + 1e-5f);
  }
}

__global__ void k_norm(const float* __restrict__ Hpw,
                       const float* __restrict__ stats,
                       half_t* __restrict__ Hn16) {
  int t = blockIdx.x * blockDim.x + threadIdx.x;
  if (t >= PWPAD * C64) return;
  float v = 0.f;
  if ((t >> 6) < PWPIX) v = (Hpw[t] - stats[0]) * stats[1];
  Hn16[t] = (half_t)v;
}

// ------------------------------------- layout converts for attention block
__global__ void k_cl2nchw(const float* __restrict__ F, float* __restrict__ X32,
                          half_t* __restrict__ X16) {
  int t = blockIdx.x * blockDim.x + threadIdx.x;   // t = c*9216 + p
  if (t >= NPIX * C64) return;
  int c = t / NPIX, p = t % NPIX;
  float v = F[(size_t)p * C64 + c];
  X32[t] = v;
  X16[t] = (half_t)v;
}

__global__ void k_nchw2cl(const float* __restrict__ X, float* __restrict__ F) {
  int t = blockIdx.x * blockDim.x + threadIdx.x;   // t = p*64 + c
  if (t >= NPIX * C64) return;
  int c = t & 63, p = t >> 6;
  F[t] = X[(size_t)c * NPIX + p];
}

// ---------------------------------------------------- att = q^T k (64x64)
__global__ void k_att(const float* __restrict__ Q, const float* __restrict__ K,
                      float* __restrict__ att) {
  __shared__ float s[256];
  int c = blockIdx.x >> 6, d = blockIdx.x & 63;
  float a = 0.f;
  for (int n = threadIdx.x; n < NPIX; n += 256)
    a += Q[(size_t)n * C64 + c] * K[(size_t)n * C64 + d];
  s[threadIdx.x] = a;
  __syncthreads();
  for (int st = 128; st > 0; st >>= 1) {
    if (threadIdx.x < st) s[threadIdx.x] += s[threadIdx.x + st];
    __syncthreads();
  }
  if (threadIdx.x == 0) att[blockIdx.x] = s[0];
}

// softmax over d for each row c, write transposed attT[d][c]
__global__ void k_softmax_t(const float* __restrict__ att, float inv,
                            float* __restrict__ attT) {
  int c = threadIdx.x;
  if (c >= 64) return;
  float mx = -1e30f;
  for (int d = 0; d < 64; ++d) mx = fmaxf(mx, att[c * 64 + d] * inv);
  float sum = 0.f;
  for (int d = 0; d < 64; ++d) sum += expf(att[c * 64 + d] * inv - mx);
  float rs = 1.0f / sum;
  for (int d = 0; d < 64; ++d)
    attT[d * 64 + c] = expf(att[c * 64 + d] * inv - mx) * rs;
}

// ================================================================ host side
extern "C" void kernel_launch(void* const* d_in, const int* in_sizes, int n_in,
                              void* d_out, int out_size, void* d_ws,
                              size_t ws_size, hipStream_t stream) {
  const float* images = (const float*)d_in[0];
  const float* ip_w = (const float*)d_in[1];
  const float* ip_b = (const float*)d_in[2];
  const float* op_w = (const float*)d_in[3];
  const float* op_b = (const float*)d_in[4];
  const float* cb_dw = (const float*)d_in[5];     // [6][64][49]
  const float* cb_pw = (const float*)d_in[6];     // [6][64][64]
  const float* cb_l1w = (const float*)d_in[7];    // [6][256][64][49]
  const float* cb_l1b = (const float*)d_in[8];    // [6][256]
  const float* cb_l2w = (const float*)d_in[9];    // [6][64][256]
  const float* cb_l2b = (const float*)d_in[10];   // [6][64]
  const float* x_qw = (const float*)d_in[11];
  const float* x_qb = (const float*)d_in[12];
  const float* x_kw = (const float*)d_in[13];
  const float* x_kb = (const float*)d_in[14];
  const float* x_vw = (const float*)d_in[15];
  const float* x_vb = (const float*)d_in[16];

  // ---- workspace carve-up
  char* wp = (char*)d_ws;
  auto alloc = [&](size_t bytes) -> void* {
    void* p = (void*)wp;
    wp += (bytes + 255) & ~(size_t)255;
    return p;
  };
  float* SB = (float*)alloc(4 * NPIX * 4);
  float* OS = (float*)alloc(4 * NPIX * 4);
  float* F0 = (float*)alloc((size_t)NPIX * C64 * 4);
  float* F1 = (float*)alloc((size_t)NPIX * C64 * 4);
  float* F2 = (float*)alloc((size_t)NPIX * C64 * 4);
  float* F3 = (float*)alloc((size_t)NPIX * C64 * 4);
  float* F4 = (float*)alloc((size_t)NPIX * C64 * 4);
  float* F5 = (float*)alloc((size_t)NPIX * C64 * 4);
  float* Hdw32 = (float*)alloc((size_t)PWPAD * C64 * 4);
  float* Hpw32 = (float*)alloc((size_t)PWPAD * C64 * 4);
  half_t* Hdw16 = (half_t*)alloc((size_t)PWPAD * C64 * 2);
  half_t* Hn16 = (half_t*)alloc((size_t)PWPAD * C64 * 2);
  half_t* Gn16 = (half_t*)alloc((size_t)NPIX * C256 * 2);
  float* Xn32 = (float*)alloc((size_t)NPIX * C64 * 4);
  half_t* Xn16 = (half_t*)alloc((size_t)NPIX * C64 * 2);
  half_t* V16 = (half_t*)alloc((size_t)NPIX * C64 * 2);
  float* On32 = (float*)alloc((size_t)NPIX * C64 * 4);
  float* Q32 = (float*)alloc((size_t)NPIX * C64 * 4);
  float* K32 = (float*)alloc((size_t)NPIX * C64 * 4);
  float* Att = (float*)alloc(64 * 64 * 4);
  float* AttT = (float*)alloc(64 * 64 * 4);
  float* Red = (float*)alloc(256 * 2 * 4);
  float* Stats = (float*)alloc(2 * 4);
  half_t* PackPW = (half_t*)alloc(4096 * 2);
  half_t* PackL1 = (half_t*)alloc((size_t)49 * 16 * 2 * 512 * 2);
  half_t* PackL2 = (half_t*)alloc(16384 * 2);
  half_t* PackQ = (half_t*)alloc(4096 * 2);
  half_t* PackK = (half_t*)alloc(4096 * 2);
  half_t* PackV = (half_t*)alloc(4096 * 2);
  half_t* PackA = (half_t*)alloc(4096 * 2);
  (void)ws_size; (void)n_in; (void)in_sizes; (void)out_size;

  // ---- ConvNeXt-style block: Y = l2(gelu(l1(norm(pw(dw(X)))))) + X [+ r2]
  auto conv_block = [&](const float* Xin, int i, const float* r2, float* Y) {
    k_dw7<<<CDIV(PWPAD * C64, 256), 256, 0, stream>>>(Xin, cb_dw + i * 64 * 49,
                                                      Hdw32, Hdw16);
    k_pack_rm<<<CDIV(4 * 2 * 512, 256), 256, 0, stream>>>(cb_pw + i * 4096, 64,
                                                          PackPW, 4, 2);
    k_gemm<<<CDIV(4 * 651, 8), 256, 0, stream>>>(
        PackPW, Hdw16, 4, 651, 2, 64, 64, PWPIX, nullptr, nullptr, nullptr,
        Hpw32, nullptr, 0);
    k_red1<<<256, 256, 0, stream>>>(Hpw32, PWPIX * C64, Red);
    k_red2<<<1, 256, 0, stream>>>(Red, (float)(PWPIX * C64), Stats);
    k_norm<<<CDIV(PWPAD * C64, 256), 256, 0, stream>>>(Hpw32, Stats, Hn16);
    k_pack_l1<<<CDIV(49 * 16 * 2 * 512, 256), 256, 0, stream>>>(
        cb_l1w + (size_t)i * 256 * 64 * 49, PackL1);
    k_l1conv<<<288, 256, 0, stream>>>(PackL1, Hn16, cb_l1b + i * 256, Gn16);
    k_pack_rm<<<CDIV(4 * 8 * 512, 256), 256, 0, stream>>>(cb_l2w + i * 64 * 256,
                                                          256, PackL2, 4, 8);
    k_gemm<<<CDIV(4 * 576, 8), 256, 0, stream>>>(
        PackL2, Gn16, 4, 576, 8, 256, 64, NPIX, cb_l2b + i * 64, Xin, r2, Y,
        nullptr, 0);
  };

  // ---- channel-attention block (flat-NCHW reinterpretation as [9216][64])
  auto xblock = [&](const float* Xin, int i, float dk, float* Y) {
    k_cl2nchw<<<CDIV(NPIX * C64, 256), 256, 0, stream>>>(Xin, Xn32, Xn16);
    k_pack_rm<<<CDIV(4096, 256), 256, 0, stream>>>(x_qw + i * 4096, 64, PackQ, 4, 2);
    k_pack_rm<<<CDIV(4096, 256), 256, 0, stream>>>(x_kw + i * 4096, 64, PackK, 4, 2);
    k_pack_rm<<<CDIV(4096, 256), 256, 0, stream>>>(x_vw + i * 4096, 64, PackV, 4, 2);
    k_gemm<<<CDIV(4 * 576, 8), 256, 0, stream>>>(PackQ, Xn16, 4, 576, 2, 64, 64,
                                                 NPIX, x_qb + i * 64, nullptr,
                                                 nullptr, Q32, nullptr, 0);
    k_gemm<<<CDIV(4 * 576, 8), 256, 0, stream>>>(PackK, Xn16, 4, 576, 2, 64, 64,
                                                 NPIX, x_kb + i * 64, nullptr,
                                                 nullptr, K32, nullptr, 0);
    k_gemm<<<CDIV(4 * 576, 8), 256, 0, stream>>>(PackV, Xn16, 4, 576, 2, 64, 64,
                                                 NPIX, x_vb + i * 64, nullptr,
                                                 nullptr, nullptr, V16, 0);
    k_att<<<4096, 256, 0, stream>>>(Q32, K32, Att);
    k_softmax_t<<<1, 64, 0, stream>>>(Att, 1.0f / sqrtf(dk), AttT);
    k_pack_rm<<<CDIV(4096, 256), 256, 0, stream>>>(AttT, 64, PackA, 4, 2);
    k_gemm<<<CDIV(4 * 576, 8), 256, 0, stream>>>(PackA, V16, 4, 576, 2, 64, 64,
                                                 NPIX, nullptr, Xn32, nullptr,
                                                 On32, nullptr, 0);
    k_nchw2cl<<<CDIV(NPIX * C64, 256), 256, 0, stream>>>(On32, Y);
  };

  // ---- forward pass
  k_dwt<<<CDIV(NPIX, 256), 256, 0, stream>>>(images, SB, SB + NPIX,
                                             SB + 2 * NPIX, SB + 3 * NPIX);
  // low-frequency path
  k_ip<<<CDIV(NPIX * C64, 256), 256, 0, stream>>>(SB, ip_w, ip_b, F0);
  conv_block(F0, 0, nullptr, F1);        // c1
  conv_block(F1, 1, F1, F2);             // c2 = cb1(c1) + c1
  xblock(F1, 0, 5.0f, F3);               // xlow
  conv_block(F3, 2, F2, F4);             // c3 = cb2(xlow) + c2
  conv_block(F4, 3, F1, F5);             // c4 = cb3(c3) + c1
  k_op<<<CDIV(NPIX, 256), 256, 0, stream>>>(F5, op_w, op_b, OS);
  // high-frequency paths (lh, hl, hh)
  for (int h = 0; h < 3; ++h) {
    const float* sb = SB + (size_t)(1 + h) * NPIX;
    k_ip<<<CDIV(NPIX * C64, 256), 256, 0, stream>>>(sb, ip_w, ip_b, F0);
    conv_block(F0, 4, nullptr, F1);      // h1
    xblock(F1, 1, 2.0f, F2);             // hx
    conv_block(F2, 5, F1, F3);           // h2 = cb5(hx) + h1
    k_op<<<CDIV(NPIX, 256), 256, 0, stream>>>(F3, op_w, op_b,
                                              OS + (size_t)(1 + h) * NPIX);
  }
  k_idwt<<<CDIV(192 * 192, 256), 256, 0, stream>>>(
      OS, OS + NPIX, OS + 2 * NPIX, OS + 3 * NPIX, images, (float*)d_out);
}
